// AlibiMultiHeadAttention_28346784154189
// MI455X (gfx1250) — matrix operand order repaired and checked by reading, not compiled, not run on hardware
//
#include <hip/hip_runtime.h>

// ---------------------------------------------------------------------------
// CDNA5 / gfx1250 ALiBi multi-head causal attention forward.
// All matmuls via v_wmma_f32_16x16x32_bf16 (wave32 WMMA).
// GEMMs use 64x64 wave tiles (16 WMMAs / k-step) for max WMMA-per-byte.
// ---------------------------------------------------------------------------

typedef __attribute__((ext_vector_type(16))) __bf16 v16bf;
typedef __attribute__((ext_vector_type(8)))  __bf16 v8bf;
typedef __attribute__((ext_vector_type(8)))  float  v8f;

#define NUM_HEADS 16
#define SEQ_T     2048
#define DIM_C     1024
#define HEAD_D    64
#define BATCH_B   4

__device__ __forceinline__ v8f zero_v8f() {
  v8f z;
#pragma unroll
  for (int i = 0; i < 8; ++i) z[i] = 0.0f;
  return z;
}

// A-fragment (16 x 32 bf16, row-major source with K contiguous).
// Lane layout: lanes 0-15 row r hold K = k0+0..15;
// lanes 16-31 hold K = k0+16..31.
__device__ __forceinline__ v16bf load_frag_a(const __bf16* __restrict__ p, int ld,
                                             int row0, int k0, int lane) {
  const int r    = lane & 15;
  const int half = lane >> 4;
  const __bf16* base = p + (size_t)(row0 + r) * ld + k0 + half * 16;
  v8bf lo = *reinterpret_cast<const v8bf*>(base);
  v8bf hi = *reinterpret_cast<const v8bf*>(base + 8);
  v16bf f;
#pragma unroll
  for (int i = 0; i < 8; ++i) { f[i] = lo[i]; f[i + 8] = hi[i]; }
  return f;
}

// B-fragment (32 x 16 bf16, column-major B == row-major B^T with K contiguous).
// Lane n (0-15): col n, K = k0+0..15; lane n+16: col n, K = k0+16..31.
__device__ __forceinline__ v16bf load_frag_b(const __bf16* __restrict__ p, int ld,
                                             int col0, int k0, int lane) {
  const int n    = lane & 15;
  const int half = lane >> 4;
  const __bf16* base = p + (size_t)(col0 + n) * ld + k0 + half * 16;
  return *reinterpret_cast<const v16bf*>(base);
}

__device__ __forceinline__ v8f wmma_bf16(v16bf a, v16bf b, v8f c) {
  return __builtin_amdgcn_wmma_f32_16x16x32_bf16(false, a, false, b, (short)0, c,
                                                 false, false);
}

// ---------------------------------------------------------------------------
// Conversion kernels
// ---------------------------------------------------------------------------
__global__ void cvt_bf16_kernel(const float* __restrict__ src,
                                __bf16* __restrict__ dst, int n) {
  for (int i = blockIdx.x * blockDim.x + threadIdx.x; i < n;
       i += gridDim.x * blockDim.x)
    dst[i] = (__bf16)src[i];
}

// src [K][N] fp32 row-major -> dst [N][K] bf16 row-major (transpose + cast)
__global__ void cvt_transpose_kernel(const float* __restrict__ src,
                                     __bf16* __restrict__ dst, int K, int N) {
  const int total = K * N;
  for (int idx = blockIdx.x * blockDim.x + threadIdx.x; idx < total;
       idx += gridDim.x * blockDim.x) {
    const int k = idx / N;
    const int n = idx - k * N;
    dst[(size_t)n * K + k] = (__bf16)src[idx];
  }
}

// ---------------------------------------------------------------------------
// QKV GEMM: xb[8192,1024] @ w_qkv  (wT = w_qkv^T [3072][1024])
// Epilogue scatters into q[B,H,T,dh], k[B,H,T,dh], vT[B,H,dh,T] (bf16).
// Block = 8 waves (2x4); wave tile 64x64; block tile 128 rows x 256 cols.
// ---------------------------------------------------------------------------
__global__ __launch_bounds__(256) void qkv_gemm_kernel(
    const __bf16* __restrict__ xb, const __bf16* __restrict__ wT,
    __bf16* __restrict__ qws, __bf16* __restrict__ kws,
    __bf16* __restrict__ vTws) {
  const int lane = threadIdx.x & 31;
  const int wid  = threadIdx.x >> 5;
  const int half = lane >> 4;
  const int row0 = blockIdx.x * 128 + (wid & 1) * 64;
  const int col0 = blockIdx.y * 256 + (wid >> 1) * 64;

  v8f acc[4][4];
#pragma unroll
  for (int i = 0; i < 4; ++i)
#pragma unroll
    for (int j = 0; j < 4; ++j) acc[i][j] = zero_v8f();

  for (int k0 = 0; k0 < DIM_C; k0 += 32) {
    __builtin_prefetch(xb + (size_t)(row0 + (lane & 15)) * DIM_C + k0 + 32, 0, 1);
    v16bf a[4], bf[4];
#pragma unroll
    for (int i = 0; i < 4; ++i)
      a[i] = load_frag_a(xb, DIM_C, row0 + i * 16, k0, lane);
#pragma unroll
    for (int j = 0; j < 4; ++j)
      bf[j] = load_frag_b(wT, DIM_C, col0 + j * 16, k0, lane);
#pragma unroll
    for (int i = 0; i < 4; ++i)
#pragma unroll
      for (int j = 0; j < 4; ++j)
        acc[i][j] = wmma_bf16(a[i], bf[j], acc[i][j]);
  }

#pragma unroll
  for (int i = 0; i < 4; ++i)
#pragma unroll
    for (int j = 0; j < 4; ++j) {
      const int c   = col0 + j * 16 + (lane & 15);
      const int sec = c >> 10;           // 0=q 1=k 2=v (1024-col sections)
      const int nn  = c & 1023;
      const int h   = nn >> 6;
      const int d   = nn & 63;
#pragma unroll
      for (int v = 0; v < 8; ++v) {
        const int m = row0 + i * 16 + v + half * 8;   // token row
        const int b = m >> 11;
        const int t = m & 2047;
        const size_t bh = (size_t)b * NUM_HEADS + h;
        const __bf16 val = (__bf16)acc[i][j][v];
        if (sec == 0)
          qws[(bh * SEQ_T + t) * HEAD_D + d] = val;
        else if (sec == 1)
          kws[(bh * SEQ_T + t) * HEAD_D + d] = val;
        else
          vTws[(bh * HEAD_D + d) * SEQ_T + t] = val;
      }
    }
}

// ---------------------------------------------------------------------------
// Attention: one wave per (b, h, 16-query tile). Transposed formulation:
//   S^T[key][q] = K_tile x Q   (both operands row-major-with-dh-contiguous)
//   softmax over keys == lane-local column reduction (+ one xor-16 shuffle)
//   O^T[d][q]  += V^T_tile x P^T   (P^T C-frag -> B layout needs one xor-16 lane exchange)
// ---------------------------------------------------------------------------
__global__ __launch_bounds__(256) void attn_kernel(
    const __bf16* __restrict__ qws, const __bf16* __restrict__ kws,
    const __bf16* __restrict__ vTws, __bf16* __restrict__ concat) {
  const int lane = threadIdx.x & 31;
  const int half = lane >> 4;
  const int gw   = (blockIdx.x * blockDim.x + threadIdx.x) >> 5;  // 0..8191
  const int qt = gw & 127;
  const int h  = (gw >> 7) & 15;
  const int b  = gw >> 11;
  const size_t bh = (size_t)b * NUM_HEADS + h;

  const __bf16* qbase = qws  + (bh * SEQ_T + (size_t)qt * 16) * HEAD_D;
  const __bf16* kbase = kws  + bh * SEQ_T * HEAD_D;
  const __bf16* vbase = vTws + bh * HEAD_D * SEQ_T;

  const float slope     = exp2f(-0.5f * (float)(h + 1));  // 1/((2^8)^((h+1)/16))
  const float inv_scale = 0.03125f;                       // 1/sqrt(1024)
  const int   qcol      = qt * 16 + (lane & 15);

  // Q as B-operand, loaded once (dh=64 -> two K=32 chunks).
  v16bf qb0 = load_frag_b(qbase, HEAD_D, 0, 0,  lane);
  v16bf qb1 = load_frag_b(qbase, HEAD_D, 0, 32, lane);

  v8f o[4];
#pragma unroll
  for (int f = 0; f < 4; ++f) o[f] = zero_v8f();
  float m = -1e30f, l = 0.0f;

  for (int jt = 0; jt <= qt; jt += 2) {
    float p0[8], p1[8];
    // ---- S^T tile jt ----
    v8f s = zero_v8f();
    s = wmma_bf16(load_frag_a(kbase, HEAD_D, jt * 16, 0,  lane), qb0, s);
    s = wmma_bf16(load_frag_a(kbase, HEAD_D, jt * 16, 32, lane), qb1, s);
    float tmax = -1e30f;
#pragma unroll
    for (int v = 0; v < 8; ++v) {
      const int key = jt * 16 + v + half * 8;
      float sc = s[v] * inv_scale + slope * (float)(key - qcol);
      sc = (key <= qcol) ? sc : -1e30f;       // causal mask
      p0[v] = sc;
      tmax = fmaxf(tmax, sc);
    }
    // ---- S^T tile jt+1 (wave-uniform predicate) ----
    const bool have1 = (jt + 1 <= qt);
    if (have1) {
      v8f s1 = zero_v8f();
      s1 = wmma_bf16(load_frag_a(kbase, HEAD_D, jt * 16 + 16, 0,  lane), qb0, s1);
      s1 = wmma_bf16(load_frag_a(kbase, HEAD_D, jt * 16 + 16, 32, lane), qb1, s1);
#pragma unroll
      for (int v = 0; v < 8; ++v) {
        const int key = jt * 16 + 16 + v + half * 8;
        float sc = s1[v] * inv_scale + slope * (float)(key - qcol);
        sc = (key <= qcol) ? sc : -1e30f;
        p1[v] = sc;
        tmax = fmaxf(tmax, sc);
      }
    } else {
#pragma unroll
      for (int v = 0; v < 8; ++v) p1[v] = -1e30f;
    }

    // ---- online softmax (per query == per lane column) ----
    tmax = fmaxf(tmax, __shfl_xor(tmax, 16, 32));
    const float mnew  = fmaxf(m, tmax);
    const float alpha = expf(m - mnew);
    float rs = 0.0f;
#pragma unroll
    for (int v = 0; v < 8; ++v) {
      p0[v] = expf(p0[v] - mnew);
      p1[v] = expf(p1[v] - mnew);
      rs += p0[v] + p1[v];
    }
    rs += __shfl_xor(rs, 16, 32);
    l = l * alpha + rs;
    m = mnew;

    // ---- build P^T B-fragment (32 keys x 16 queries) ----
    // lanes<16 need partner's p0 (keys 8..15); lanes>=16 need partner's p1.
    v16bf pb;
#pragma unroll
    for (int v = 0; v < 8; ++v) {
      const float send = (half == 0) ? p1[v] : p0[v];
      const float recv = __shfl_xor(send, 16, 32);
      const float e_lo = (half == 0) ? p0[v] : recv;  // elements 0..7
      const float e_hi = (half == 0) ? recv : p1[v];  // elements 8..15
      pb[v]     = (__bf16)e_lo;
      pb[v + 8] = (__bf16)e_hi;
    }

    // ---- rescale accumulator, then O^T += V^T x P^T ----
#pragma unroll
    for (int f = 0; f < 4; ++f)
#pragma unroll
      for (int v = 0; v < 8; ++v) o[f][v] *= alpha;
#pragma unroll
    for (int f = 0; f < 4; ++f) {
      v16bf va = load_frag_a(vbase, SEQ_T, f * 16, jt * 16, lane);
      o[f] = wmma_bf16(va, pb, o[f]);
    }
  }

  // ---- epilogue: concat[b, t, h*64 + d] = O^T[d][q] / l ----
  const float invl = 1.0f / l;
  const int trow = b * SEQ_T + qt * 16 + (lane & 15);
#pragma unroll
  for (int f = 0; f < 4; ++f)
#pragma unroll
    for (int v = 0; v < 8; ++v) {
      const int d = f * 16 + v + half * 8;
      concat[(size_t)trow * DIM_C + h * HEAD_D + d] = (__bf16)(o[f][v] * invl);
    }
}

// ---------------------------------------------------------------------------
// Output projection: concat[8192,1024] @ w_o (woT = w_o^T [1024][1024]) -> fp32
// Block = 8 waves (2x4); wave tile 64x64; block tile 128 rows x 256 cols.
// ---------------------------------------------------------------------------
__global__ __launch_bounds__(256) void out_gemm_kernel(
    const __bf16* __restrict__ concat, const __bf16* __restrict__ woT,
    float* __restrict__ out) {
  const int lane = threadIdx.x & 31;
  const int wid  = threadIdx.x >> 5;
  const int half = lane >> 4;
  const int row0 = blockIdx.x * 128 + (wid & 1) * 64;
  const int col0 = blockIdx.y * 256 + (wid >> 1) * 64;

  v8f acc[4][4];
#pragma unroll
  for (int i = 0; i < 4; ++i)
#pragma unroll
    for (int j = 0; j < 4; ++j) acc[i][j] = zero_v8f();

  for (int k0 = 0; k0 < DIM_C; k0 += 32) {
    __builtin_prefetch(concat + (size_t)(row0 + (lane & 15)) * DIM_C + k0 + 32, 0, 1);
    v16bf a[4], bf[4];
#pragma unroll
    for (int i = 0; i < 4; ++i)
      a[i] = load_frag_a(concat, DIM_C, row0 + i * 16, k0, lane);
#pragma unroll
    for (int j = 0; j < 4; ++j)
      bf[j] = load_frag_b(woT, DIM_C, col0 + j * 16, k0, lane);
#pragma unroll
    for (int i = 0; i < 4; ++i)
#pragma unroll
      for (int j = 0; j < 4; ++j)
        acc[i][j] = wmma_bf16(a[i], bf[j], acc[i][j]);
  }

#pragma unroll
  for (int i = 0; i < 4; ++i)
#pragma unroll
    for (int j = 0; j < 4; ++j) {
      const int c = col0 + j * 16 + (lane & 15);
#pragma unroll
      for (int v = 0; v < 8; ++v) {
        const int mrow = row0 + i * 16 + v + half * 8;
        out[(size_t)mrow * DIM_C + c] = acc[i][j][v];
      }
    }
}

// ---------------------------------------------------------------------------
// Host launcher
// ---------------------------------------------------------------------------
extern "C" void kernel_launch(void* const* d_in, const int* in_sizes, int n_in,
                              void* d_out, int out_size, void* d_ws, size_t ws_size,
                              hipStream_t stream) {
  (void)in_sizes; (void)n_in; (void)out_size; (void)ws_size;
  const float* x     = (const float*)d_in[0];   // [4,2048,1024]
  const float* w_qkv = (const float*)d_in[1];   // [1024,3072]
  const float* w_o   = (const float*)d_in[2];   // [1024,1024]
  float* out = (float*)d_out;                   // [4,2048,1024] fp32

  const size_t MT = (size_t)BATCH_B * SEQ_T;          // 8192 rows
  char* ws = (char*)d_ws;
  __bf16* xb   = (__bf16*)ws;  ws += MT * DIM_C * 2;                 // 16 MB
  __bf16* wqT  = (__bf16*)ws;  ws += (size_t)3 * DIM_C * DIM_C * 2;  //  6 MB
  __bf16* woT  = (__bf16*)ws;  ws += (size_t)DIM_C * DIM_C * 2;      //  2 MB
  __bf16* qws  = (__bf16*)ws;  ws += MT * DIM_C * 2;                 // 16 MB
  __bf16* kws  = (__bf16*)ws;  ws += MT * DIM_C * 2;                 // 16 MB
  __bf16* vT   = (__bf16*)ws;  ws += MT * DIM_C * 2;                 // 16 MB
  __bf16* cc   = (__bf16*)ws;                                        // 16 MB

  cvt_bf16_kernel<<<2048, 256, 0, stream>>>(x, xb, (int)(MT * DIM_C));
  cvt_transpose_kernel<<<3072, 256, 0, stream>>>(w_qkv, wqT, DIM_C, 3 * DIM_C);
  cvt_transpose_kernel<<<1024, 256, 0, stream>>>(w_o, woT, DIM_C, DIM_C);

  // 8192x3072 GEMM: block tile 128x256 -> grid (64, 12)
  qkv_gemm_kernel<<<dim3(64, 12), 256, 0, stream>>>(xb, wqT, qws, kws, vT);

  // 8192 waves (one per b,h,q-tile), 8 waves per block -> 1024 blocks
  attn_kernel<<<1024, 256, 0, stream>>>(qws, kws, vT, cc);

  // 8192x1024 GEMM: block tile 128x256 -> grid (64, 4)
  out_gemm_kernel<<<dim3(64, 4), 256, 0, stream>>>(cc, woT, out);
}
